// MixConvNet_19885698581028
// MI455X (gfx1250) — matrix operand order repaired and checked by reading, not compiled, not run on hardware
//
#include <hip/hip_runtime.h>
#include <cstddef>

typedef __attribute__((ext_vector_type(16))) _Float16 v16h;
typedef __attribute__((ext_vector_type(8)))  _Float16 v8h;
typedef __attribute__((ext_vector_type(4)))  _Float16 v4h;
typedef __attribute__((ext_vector_type(8)))  float    v8f;

union HFrag { v16h v; v8h h[2]; };

#define EPSF 1e-5f

// ---------------------------------------------------------------------------
// Weight prep: f32 -> f16, with im2col-friendly K ordering for the big conv.
//   Aw[m][r*224 + cd] = w2a[m][cd][kh][kw], r = kh*3+kw   (K = 2016 = 63*32)
//   Wq[0:64]=m_w1, [64:128]=m_w2, [128:192]=m_w3  (rows x 64)
// ---------------------------------------------------------------------------
__global__ void mcn_prep(const float* __restrict__ w2a,
                         const float* __restrict__ mw1,
                         const float* __restrict__ mw2,
                         const float* __restrict__ mw3,
                         _Float16* __restrict__ Aw,
                         _Float16* __restrict__ Wq) {
  int i = blockIdx.x * 256 + threadIdx.x;
  if (i < 64 * 2016) {
    int m = i / 2016, k = i % 2016;
    int r = k / 224, cd = k % 224;
    Aw[i] = (_Float16)w2a[(m * 224 + cd) * 9 + r];
  }
  if (i < 192 * 64) {
    int m = i >> 6, c = i & 63;
    float v = (m < 64) ? mw1[m * 64 + c]
                       : (m < 128) ? mw2[(m - 64) * 64 + c]
                                   : mw3[(m - 128) * 64 + c];
    Wq[i] = (_Float16)v;
  }
}

// ---------------------------------------------------------------------------
// conv3d(1->8, 3x3x3, VALID) + BN + ReLU, fused with (C,D)->C*D reshape.
// Output directly in f16 (only consumer is the WMMA im2col).
// y224h layout: [b][cd=c*28+d][34][34]
// ---------------------------------------------------------------------------
__global__ void mcn_conv3(const float* __restrict__ x1,
                          const float* __restrict__ w3, const float* __restrict__ b3,
                          const float* __restrict__ g3, const float* __restrict__ be3,
                          const float* __restrict__ m3, const float* __restrict__ v3,
                          _Float16* __restrict__ y224h) {
  int cd = blockIdx.x;          // 0..223
  int b  = blockIdx.y;          // 0..7
  int c = cd / 28, d = cd % 28;
  float w[27];
#pragma unroll
  for (int t = 0; t < 27; ++t) w[t] = w3[c * 27 + t];
  float sc = g3[c] * rsqrtf(v3[c] + EPSF);
  float sh = be3[c] - m3[c] * sc;
  float bias = b3[c];
  const float* xb = x1 + (size_t)b * 30 * 36 * 36;
  _Float16* yb = y224h + (size_t)(b * 224 + cd) * 1156;
  for (int p = threadIdx.x; p < 1156; p += blockDim.x) {
    int h = p / 34, ww = p % 34;
    float acc = bias;
#pragma unroll
    for (int kd = 0; kd < 3; ++kd)
#pragma unroll
      for (int kh = 0; kh < 3; ++kh)
#pragma unroll
        for (int kw = 0; kw < 3; ++kw)
          acc += xb[((d + kd) * 36 + (h + kh)) * 36 + (ww + kw)] * w[(kd * 3 + kh) * 3 + kw];
    float r = sc * acc + sh;
    yb[p] = (_Float16)(r > 0.f ? r : 0.f);
  }
}

// ---------------------------------------------------------------------------
// conv2d 224->64 3x3 VALID as implicit GEMM with WMMA f16 (M=64,N=8192,K=2016)
// + fused bias + BN + ReLU, f16 output.  256 thr (8 waves), tile 64x64.
// A staged as one b128 per thread; B gathered 4 K-values -> one ds b64.
// ---------------------------------------------------------------------------
__global__ __launch_bounds__(256) void mcn_conv2(
    const _Float16* __restrict__ y224h, const _Float16* __restrict__ Aw,
    const float* __restrict__ b2a, const float* __restrict__ g2a,
    const float* __restrict__ be2a, const float* __restrict__ m2a,
    const float* __restrict__ v2a, _Float16* __restrict__ y64h) {
  __shared__ alignas(64) _Float16 Alds[64 * 32];   // [m][kc]
  __shared__ alignas(64) _Float16 Blds[64 * 32];   // [pixel][kc]  (B^T)
  int b  = blockIdx.y;
  int p0 = blockIdx.x * 64;
  int tid  = threadIdx.x;
  int wv   = tid >> 5;
  int lane = tid & 31;
  int mi  = wv >> 1;          // 0..3
  int ni0 = (wv & 1) * 2;     // 0 or 2
  v8f acc0 = {}; v8f acc1 = {};
  const _Float16* yb = y224h + (size_t)b * 224 * 1156;
  int hlf = lane >> 4;
  int lm  = lane & 15;
  int am  = tid >> 2;            // A staging: row
  int aoff = (tid & 3) * 8;      // A staging: 8-f16 offset
  for (int t = 0; t < 63; ++t) {
    int r = t / 7;
    int kh = r / 3, kw = r % 3;
    int c0 = (t % 7) * 32;
    // A: 2048 f16 = 256 threads x one b128
    *(v8h*)&Alds[am * 32 + aoff] =
        *(const v8h*)(Aw + (size_t)am * 2016 + t * 32 + aoff);
    // B: 2048 f16 = 512 items of 4 K-values (channel-gather) each
    for (int idx = tid; idx < 512; idx += 256) {
      int kc = (idx >> 6) * 4, pl = idx & 63;
      int p = p0 + pl;
      int h = p >> 5, w = p & 31;
      const _Float16* src = yb + (size_t)(c0 + kc) * 1156 + (h + kh) * 34 + (w + kw);
      v4h pk;
#pragma unroll
      for (int j = 0; j < 4; ++j) pk[j] = src[(size_t)j * 1156];
      *(v4h*)&Blds[pl * 32 + kc] = pk;
    }
    if (t + 1 < 63) __builtin_prefetch(Aw + (size_t)(t + 1) * 32, 0, 0);
    __syncthreads();
    HFrag af, bf0, bf1;
    const _Float16* arow = &Alds[(mi * 16 + lm) * 32];
    af.h[0] = *(const v8h*)(arow + hlf * 16);
    af.h[1] = *(const v8h*)(arow + hlf * 16 + 8);
    bf0.v = *(const v16h*)&Blds[(ni0 * 16 + lm) * 32 + hlf * 16];
    bf1.v = *(const v16h*)&Blds[((ni0 + 1) * 16 + lm) * 32 + hlf * 16];
    acc0 = __builtin_amdgcn_wmma_f32_16x16x32_f16(false, af.v, false, bf0.v, (short)0, acc0, false, false);
    acc1 = __builtin_amdgcn_wmma_f32_16x16x32_f16(false, af.v, false, bf1.v, (short)0, acc1, false, false);
    __syncthreads();
  }
  // Epilogue from C/D layout: lane 0-15 -> N=lane, M=j ; lane 16-31 -> M=8+j
#pragma unroll
  for (int j = 0; j < 8; ++j) {
    int ch = mi * 16 + hlf * 8 + j;
    float sc = g2a[ch] * rsqrtf(v2a[ch] + EPSF);
    float sh = be2a[ch] - m2a[ch] * sc;
    float v0 = sc * (acc0[j] + b2a[ch]) + sh; v0 = v0 > 0.f ? v0 : 0.f;
    float v1 = sc * (acc1[j] + b2a[ch]) + sh; v1 = v1 > 0.f ? v1 : 0.f;
    y64h[(size_t)(b * 64 + ch) * 1024 + p0 + ni0 * 16 + lm] = (_Float16)v0;
    y64h[(size_t)(b * 64 + ch) * 1024 + p0 + (ni0 + 1) * 16 + lm] = (_Float16)v1;
  }
}

// ---------------------------------------------------------------------------
// conv2d 1->64 3x3 VALID + BN + ReLU (x2 branch; K=9, direct), f16 output.
// ---------------------------------------------------------------------------
__global__ void mcn_conv1(const float* __restrict__ x2,
                          const float* __restrict__ w2b, const float* __restrict__ b2b,
                          const float* __restrict__ g2b, const float* __restrict__ be2b,
                          const float* __restrict__ m2b, const float* __restrict__ v2b,
                          _Float16* __restrict__ z64h) {
  int ch = blockIdx.x;  int b = blockIdx.y;
  float w[9];
#pragma unroll
  for (int t = 0; t < 9; ++t) w[t] = w2b[ch * 9 + t];
  float sc = g2b[ch] * rsqrtf(v2b[ch] + EPSF);
  float sh = be2b[ch] - m2b[ch] * sc;
  float bias = b2b[ch];
  const float* xb = x2 + (size_t)b * 1156;
  _Float16* zb = z64h + (size_t)(b * 64 + ch) * 1024;
  for (int p = threadIdx.x; p < 1024; p += blockDim.x) {
    int h = p >> 5, ww = p & 31;
    float acc = bias;
#pragma unroll
    for (int kh = 0; kh < 3; ++kh)
#pragma unroll
      for (int kw = 0; kw < 3; ++kw)
        acc += xb[(h + kh) * 34 + (ww + kw)] * w[kh * 3 + kw];
    float r = sc * acc + sh;
    zb[p] = (_Float16)(r > 0.f ? r : 0.f);
  }
}

// ---------------------------------------------------------------------------
// Fused q/k/v projection: GEMM M=192 (q|k|v), K=64, N=8192 via WMMA + bias.
// qkv layout (f32): [b][m=part*64+head*8+d][1024]
// ---------------------------------------------------------------------------
__global__ __launch_bounds__(256) void mcn_qkv(
    const _Float16* __restrict__ xin, const _Float16* __restrict__ Wq,
    const float* __restrict__ b1, const float* __restrict__ b2,
    const float* __restrict__ b3, float* __restrict__ qkv) {
  __shared__ alignas(64) _Float16 Alds[192 * 64];   // [m][c]
  __shared__ alignas(64) _Float16 Blds[64 * 64];    // [pixel][c]
  int b = blockIdx.y; int p0 = blockIdx.x * 64;
  int tid = threadIdx.x; int wv = tid >> 5; int lane = tid & 31;
  int hlf = lane >> 4, lm = lane & 15;
  // A: 12288 f16 as 1536 b128 moves
  for (int idx = tid; idx < 1536; idx += 256)
    *(v8h*)&Alds[idx * 8] = *(const v8h*)(Wq + (size_t)idx * 8);
  // B: 4096 f16 as 1024 items of 4 channel-gathered values
  const _Float16* xb = xin + (size_t)b * 64 * 1024;
  for (int idx = tid; idx < 1024; idx += 256) {
    int pl = idx >> 4, c = (idx & 15) * 4;
    v4h pk;
#pragma unroll
    for (int j = 0; j < 4; ++j) pk[j] = xb[(size_t)(c + j) * 1024 + p0 + pl];
    *(v4h*)&Blds[pl * 64 + c] = pk;
  }
  __syncthreads();
  v8f acc[6] = {};
#pragma unroll
  for (int kc = 0; kc < 64; kc += 32) {
#pragma unroll
    for (int i = 0; i < 6; ++i) {
      int tile = wv * 6 + i;
      int mi = tile >> 2, ni = tile & 3;
      HFrag af, bf;
      const _Float16* arow = &Alds[(mi * 16 + lm) * 64 + kc];
      af.h[0] = *(const v8h*)(arow + hlf * 16);
      af.h[1] = *(const v8h*)(arow + hlf * 16 + 8);
      bf.v = *(const v16h*)&Blds[(ni * 16 + lm) * 64 + kc + hlf * 16];
      acc[i] = __builtin_amdgcn_wmma_f32_16x16x32_f16(false, af.v, false, bf.v, (short)0, acc[i], false, false);
    }
  }
  float* qb = qkv + (size_t)b * 192 * 1024;
#pragma unroll
  for (int i = 0; i < 6; ++i) {
    int tile = wv * 6 + i; int mi = tile >> 2, ni = tile & 3;
#pragma unroll
    for (int j = 0; j < 8; ++j) {
      int m = mi * 16 + hlf * 8 + j;
      float bias = (m < 64) ? b1[m] : (m < 128 ? b2[m - 64] : b3[m - 128]);
      qb[(size_t)m * 1024 + p0 + ni * 16 + lm] = acc[i][j] + bias;
    }
  }
}

// ---------------------------------------------------------------------------
// MACT 7x7 local attention (reflect-pad).
// pe enters linearly:  sum_d q_d*(pe_p - pe_pr)_d = qx*(lw-lwr) + qy*(lh-lhr),
// so per tap we do one ds_load_b128 of k (f16 [pixel][d]) + 8 FMA + 2 FMA.
// k and v staged as f16 in LDS (32 KB total).
// ---------------------------------------------------------------------------
__global__ __launch_bounds__(256) void mcn_att(const float* __restrict__ qkv,
                                               const float* __restrict__ wp,
                                               const float* __restrict__ bp,
                                               float* __restrict__ attout) {
  __shared__ alignas(16) _Float16 klds[1024 * 8];   // [p][d]
  __shared__ alignas(16) _Float16 vlds[1024 * 8];   // [p][d]
  int bh = blockIdx.x;                 // b*8 + head
  int b = bh >> 3, head = bh & 7;
  const float* qbase = qkv + (size_t)(b * 192 + head * 8) * 1024;
  const float* kbase = qkv + (size_t)(b * 192 + 64 + head * 8) * 1024;
  const float* vbase = qkv + (size_t)(b * 192 + 128 + head * 8) * 1024;
  int tid = threadIdx.x;
  for (int idx = tid; idx < 8192; idx += 256) {
    int d = idx >> 10, p = idx & 1023;
    klds[p * 8 + d] = (_Float16)kbase[idx];
    vlds[p * 8 + d] = (_Float16)vbase[idx];
  }
  __syncthreads();
  float wpx[8], wpy[8];
#pragma unroll
  for (int d = 0; d < 8; ++d) { wpx[d] = wp[2 * d]; wpy[d] = wp[2 * d + 1]; }
  (void)bp;                                  // cancels in pe_p - pe_pr
  const float scaling = 0.35355339059327373f;   // HD^-0.5
  const float lstep = 2.f / 31.f;
  for (int pi = 0; pi < 4; ++pi) {
    int p = pi * 256 + tid;
    int h = p >> 5, w = p & 31;
    float q[8], qx = 0.f, qy = 0.f;
#pragma unroll
    for (int d = 0; d < 8; ++d) {
      q[d] = qbase[d * 1024 + p] * scaling;
      qx += q[d] * wpx[d];
      qy += q[d] * wpy[d];
    }
    float s[49]; float mx = -1e30f;
#pragma unroll
    for (int idx = 0; idx < 49; ++idx) {
      int ki = idx / 7, kj = idx % 7;
      int hr = h + ki - 3; hr = hr < 0 ? -hr : (hr > 31 ? 62 - hr : hr);
      int wr = w + kj - 3; wr = wr < 0 ? -wr : (wr > 31 ? 62 - wr : wr);
      int pr = hr * 32 + wr;
      v8h kv = *(const v8h*)&klds[pr * 8];
      float acc = qx * ((w - wr) * lstep) + qy * ((h - hr) * lstep);
#pragma unroll
      for (int d = 0; d < 8; ++d) acc += q[d] * (float)kv[d];
      s[idx] = acc; mx = fmaxf(mx, acc);
    }
    float sum = 0.f;
#pragma unroll
    for (int idx = 0; idx < 49; ++idx) { s[idx] = __expf(s[idx] - mx); sum += s[idx]; }
    float inv = 1.f / sum;
    float out[8] = {};
#pragma unroll
    for (int idx = 0; idx < 49; ++idx) {
      int ki = idx / 7, kj = idx % 7;
      int hr = h + ki - 3; hr = hr < 0 ? -hr : (hr > 31 ? 62 - hr : hr);
      int wr = w + kj - 3; wr = wr < 0 ? -wr : (wr > 31 ? 62 - wr : wr);
      int pr = hr * 32 + wr;
      v8h vv = *(const v8h*)&vlds[pr * 8];
      float a = s[idx] * inv;
#pragma unroll
      for (int d = 0; d < 8; ++d) out[d] += a * (float)vv[d];
    }
    float* ob = attout + (size_t)(b * 64 + head * 8) * 1024;
#pragma unroll
    for (int d = 0; d < 8; ++d) ob[d * 1024 + p] = out[d];
  }
}

// ---------------------------------------------------------------------------
// MACT conv path: m_fc mixing (24->9) + grouped 3x3 conv (groups=HD),
// fused with rate1*att + rate2*conv.  One block per (b, d=group).
// ---------------------------------------------------------------------------
__global__ __launch_bounds__(256) void mcn_mact_conv(
    const float* __restrict__ qkv, const float* __restrict__ attout,
    const float* __restrict__ fc, const float* __restrict__ depw,
    const float* __restrict__ depb, const float* __restrict__ rate1p,
    const float* __restrict__ rate2p, float* __restrict__ mout) {
  __shared__ float fpl[9 * 1024];
  __shared__ float fcs[9 * 24];
  int bd = blockIdx.x;  int b = bd >> 3, d = bd & 7;
  int tid = threadIdx.x;
  for (int i = tid; i < 216; i += 256) fcs[i] = fc[i];
  __syncthreads();
  const float* qb = qkv + (size_t)b * 192 * 1024;
  for (int idx = tid; idx < 9 * 1024; idx += 256) {
    int o = idx >> 10, p = idx & 1023;
    float acc = 0.f;
#pragma unroll
    for (int s = 0; s < 24; ++s) {
      int part = s >> 3; int hd = s & 7;
      acc += fcs[o * 24 + s] * qb[(size_t)(part * 64 + hd * 8 + d) * 1024 + p];
    }
    fpl[idx] = acc;
  }
  __syncthreads();
  float r1 = rate1p[0], r2 = rate2p[0];
  for (int idx = tid; idx < 8 * 1024; idx += 256) {
    int oc = idx >> 10, p = idx & 1023;
    int h = p >> 5, w = p & 31;
    int c = d * 8 + oc;
    float acc = depb[c];
#pragma unroll
    for (int ic = 0; ic < 9; ++ic) {
#pragma unroll
      for (int kh = 0; kh < 3; ++kh) {
        int hh = h + kh - 1; if (hh < 0 || hh > 31) continue;
#pragma unroll
        for (int kw = 0; kw < 3; ++kw) {
          int wwp = w + kw - 1; if (wwp < 0 || wwp > 31) continue;
          acc += fpl[ic * 1024 + hh * 32 + wwp] * depw[(c * 9 + ic) * 9 + kh * 3 + kw];
        }
      }
    }
    size_t oi = (size_t)(b * 64 + c) * 1024 + p;
    mout[oi] = r1 * attout[oi] + r2 * acc;
  }
}

// ---------------------------------------------------------------------------
// Tokenizer, algebraically reduced: T[l] = (sum_n softmax_n(t wA_l)[n] t[n]) wV
// One block per (b, l).  Also builds s = concat(cls, T) + pos.
// ---------------------------------------------------------------------------
__global__ __launch_bounds__(256) void mcn_token(
    const float* __restrict__ mout, const float* __restrict__ wA,
    const float* __restrict__ wV, const float* __restrict__ pos,
    const float* __restrict__ cls, float* __restrict__ sbuf) {
  __shared__ float logits[1024];
  __shared__ float red[32];
  __shared__ float upart[4][64];
  int l = blockIdx.x; int b = blockIdx.y;
  int tid = threadIdx.x;
  const float* tb = mout + (size_t)b * 64 * 1024;   // t[n][c] = tb[c*1024+n]
  const float* wa = wA + l * 64;
  for (int n = tid; n < 1024; n += 256) {
    float acc = 0.f;
#pragma unroll
    for (int c = 0; c < 64; ++c) acc += tb[(size_t)c * 1024 + n] * wa[c];
    logits[n] = acc;
  }
  __syncthreads();
  float mx = -1e30f;
  for (int n = tid; n < 1024; n += 256) mx = fmaxf(mx, logits[n]);
  for (int off = 16; off; off >>= 1) mx = fmaxf(mx, __shfl_down(mx, off, 32));
  if ((tid & 31) == 0) red[tid >> 5] = mx;
  __syncthreads();
  if (tid == 0) { float m = red[0]; for (int i = 1; i < 8; ++i) m = fmaxf(m, red[i]); red[0] = m; }
  __syncthreads();
  mx = red[0];
  float sm = 0.f;
  for (int n = tid; n < 1024; n += 256) { float e = __expf(logits[n] - mx); logits[n] = e; sm += e; }
  for (int off = 16; off; off >>= 1) sm += __shfl_down(sm, off, 32);
  if ((tid & 31) == 0) red[8 + (tid >> 5)] = sm;
  __syncthreads();
  if (tid == 0) { float s = 0.f; for (int i = 0; i < 8; ++i) s += red[8 + i]; red[16] = 1.f / s; }
  __syncthreads();
  float inv = red[16];
  int c = tid & 63; int seg = tid >> 6;              // 4 segments of 256 tokens
  float acc = 0.f;
  for (int n = seg * 256; n < seg * 256 + 256; ++n) acc += logits[n] * tb[(size_t)c * 1024 + n];
  upart[seg][c] = acc;
  __syncthreads();
  if (tid < 64) {
    float u = (upart[0][tid] + upart[1][tid] + upart[2][tid] + upart[3][tid]) * inv;
    upart[0][tid] = u;
  }
  __syncthreads();
  if (tid < 64) {
    int k = tid;
    float acc2 = 0.f;
#pragma unroll
    for (int cc = 0; cc < 64; ++cc) acc2 += upart[0][cc] * wV[cc * 64 + k];
    sbuf[(b * 5 + 1 + l) * 64 + k] = acc2 + pos[(1 + l) * 64 + k];
    if (l == 0) sbuf[(b * 5) * 64 + k] = cls[k] + pos[k];
  }
}

// ---------------------------------------------------------------------------
// MCGF (two 5-token MHA blocks, twice) + LayerNorm + linear head. Per-batch.
// ---------------------------------------------------------------------------
__device__ void mcn_proj(const float* X, const float* W, const float* bias,
                         const float* add, float* Y, int tid, int nt) {
  for (int idx = tid; idx < 320; idx += nt) {
    int n = idx >> 6, o = idx & 63;
    float acc = bias ? bias[o] : 0.f;
    if (add) acc += add[idx];
    const float* x = X + n * 64; const float* w = W + o * 64;
#pragma unroll
    for (int i = 0; i < 64; ++i) acc += x[i] * w[i];
    Y[idx] = acc;
  }
}

__device__ void mcn_mha(const float* qx, const float* kx, const float* vx,
                        const float* wq, const float* wk, const float* wv,
                        const float* wo, const float* bo, const float* add,
                        float* Q, float* K, float* V, float* O, float* Y,
                        int tid, int nt) {
  mcn_proj(qx, wq, nullptr, nullptr, Q, tid, nt);
  mcn_proj(kx, wk, nullptr, nullptr, K, tid, nt);
  mcn_proj(vx, wv, nullptr, nullptr, V, tid, nt);
  __syncthreads();
  if (tid < 40) {                       // 8 heads x 5 query rows
    int h = tid / 5, i = tid % 5;
    const float sc = 0.35355339059327373f;
    float lg[5]; float mx = -1e30f;
#pragma unroll
    for (int j = 0; j < 5; ++j) {
      float a = 0.f;
#pragma unroll
      for (int d = 0; d < 8; ++d) a += Q[i * 64 + h * 8 + d] * K[j * 64 + h * 8 + d];
      lg[j] = a * sc; mx = fmaxf(mx, lg[j]);
    }
    float s = 0.f;
#pragma unroll
    for (int j = 0; j < 5; ++j) { lg[j] = __expf(lg[j] - mx); s += lg[j]; }
    float inv = 1.f / s;
#pragma unroll
    for (int d = 0; d < 8; ++d) {
      float a = 0.f;
#pragma unroll
      for (int j = 0; j < 5; ++j) a += lg[j] * V[j * 64 + h * 8 + d];
      O[i * 64 + h * 8 + d] = a * inv;
    }
  }
  __syncthreads();
  mcn_proj(O, wo, bo, add, Y, tid, nt);
  __syncthreads();
}

__global__ __launch_bounds__(128) void mcn_head(
    const float* __restrict__ s1g, const float* __restrict__ s2g,
    const float* __restrict__ wq, const float* __restrict__ wk,
    const float* __restrict__ wv, const float* __restrict__ wo,
    const float* __restrict__ bo, const float* __restrict__ lng,
    const float* __restrict__ lnb, const float* __restrict__ hw,
    const float* __restrict__ hb, float* __restrict__ out) {
  __shared__ float SA[320], SB[320], Q[320], K[320], V[320], O[320], M1[320], R[320];
  __shared__ float hacc[2][64];
  int b = blockIdx.x; int tid = threadIdx.x; const int nt = 128;
  for (int i = tid; i < 320; i += nt) { SA[i] = s1g[b * 320 + i]; SB[i] = s2g[b * 320 + i]; }
  __syncthreads();
  // o1 = mcgf(s1, s2):  M1 = s1 + mha(s1,s2,s2);  R = mha(M1,M1,s1)
  mcn_mha(SA, SB, SB, wq, wk, wv, wo, bo, SA, Q, K, V, O, M1, tid, nt);
  mcn_mha(M1, M1, SA, wq, wk, wv, wo, bo, nullptr, Q, K, V, O, R, tid, nt);
  if (tid < 64) hacc[0][tid] = R[tid];
  __syncthreads();
  // o2 = mcgf(s2, s1)
  mcn_mha(SB, SA, SA, wq, wk, wv, wo, bo, SB, Q, K, V, O, M1, tid, nt);
  mcn_mha(M1, M1, SB, wq, wk, wv, wo, bo, nullptr, Q, K, V, O, R, tid, nt);
  if (tid < 64) hacc[1][tid] = R[tid];
  __syncthreads();
  if (tid == 0) {
    float res[6];
    for (int j = 0; j < 6; ++j) res[j] = 2.f * hb[j];
    for (int u = 0; u < 2; ++u) {
      float mu = 0.f; for (int c = 0; c < 64; ++c) mu += hacc[u][c]; mu *= (1.f / 64.f);
      float var = 0.f;
      for (int c = 0; c < 64; ++c) { float dd = hacc[u][c] - mu; var += dd * dd; }
      var *= (1.f / 64.f);
      float inv = rsqrtf(var + EPSF);
      for (int j = 0; j < 6; ++j) {
        float a = 0.f;
        for (int c = 0; c < 64; ++c)
          a += ((hacc[u][c] - mu) * inv * lng[c] + lnb[c]) * hw[j * 64 + c];
        res[j] += a;
      }
    }
    for (int j = 0; j < 6; ++j) out[b * 6 + j] = res[j];
  }
}

// ---------------------------------------------------------------------------
extern "C" void kernel_launch(void* const* d_in, const int* in_sizes, int n_in,
                              void* d_out, int out_size, void* d_ws, size_t ws_size,
                              hipStream_t stream) {
  (void)in_sizes; (void)n_in; (void)out_size; (void)ws_size;
  const float* x1   = (const float*)d_in[0];
  const float* x2   = (const float*)d_in[1];
  const float* w3   = (const float*)d_in[2];
  const float* b3   = (const float*)d_in[3];
  const float* g3   = (const float*)d_in[4];
  const float* be3  = (const float*)d_in[5];
  const float* m3   = (const float*)d_in[6];
  const float* v3   = (const float*)d_in[7];
  const float* w2a  = (const float*)d_in[8];
  const float* b2a  = (const float*)d_in[9];
  const float* g2a  = (const float*)d_in[10];
  const float* be2a = (const float*)d_in[11];
  const float* m2a  = (const float*)d_in[12];
  const float* v2a  = (const float*)d_in[13];
  const float* w2b  = (const float*)d_in[14];
  const float* b2b  = (const float*)d_in[15];
  const float* g2b  = (const float*)d_in[16];
  const float* be2b = (const float*)d_in[17];
  const float* m2b  = (const float*)d_in[18];
  const float* v2b  = (const float*)d_in[19];
  const float* m_w1 = (const float*)d_in[20];
  const float* m_b1 = (const float*)d_in[21];
  const float* m_w2 = (const float*)d_in[22];
  const float* m_b2 = (const float*)d_in[23];
  const float* m_w3 = (const float*)d_in[24];
  const float* m_b3 = (const float*)d_in[25];
  const float* m_wp = (const float*)d_in[26];
  const float* m_bp = (const float*)d_in[27];
  const float* m_fc = (const float*)d_in[28];
  const float* m_depw = (const float*)d_in[29];
  const float* m_depb = (const float*)d_in[30];
  const float* m_rate1 = (const float*)d_in[31];
  const float* m_rate2 = (const float*)d_in[32];
  const float* token_wA = (const float*)d_in[33];
  const float* token_wV = (const float*)d_in[34];
  const float* pos  = (const float*)d_in[35];
  const float* cls  = (const float*)d_in[36];
  const float* c_wq = (const float*)d_in[37];
  const float* c_wk = (const float*)d_in[38];
  const float* c_wv = (const float*)d_in[39];
  const float* c_wo = (const float*)d_in[40];
  const float* c_bo = (const float*)d_in[41];
  const float* ln_g = (const float*)d_in[42];
  const float* ln_b = (const float*)d_in[43];
  const float* head_w = (const float*)d_in[44];
  const float* head_b = (const float*)d_in[45];

  size_t off = 0;
  char* ws = (char*)d_ws;
  auto take = [&](size_t bytes) -> void* {
    void* p = ws + off;
    off += (bytes + 255) & ~(size_t)255;
    return p;
  };
  _Float16* y224h = (_Float16*)take(sizeof(_Float16) * 8 * 224 * 1156);
  _Float16* Aw    = (_Float16*)take(sizeof(_Float16) * 64 * 2016);
  _Float16* Wq    = (_Float16*)take(sizeof(_Float16) * 192 * 64);
  _Float16* y64h  = (_Float16*)take(sizeof(_Float16) * 8 * 64 * 1024);
  _Float16* z64h  = (_Float16*)take(sizeof(_Float16) * 8 * 64 * 1024);
  float* qkv1 = (float*)take(sizeof(float) * 8 * 192 * 1024);
  float* qkv2 = (float*)take(sizeof(float) * 8 * 192 * 1024);
  float* att1 = (float*)take(sizeof(float) * 8 * 64 * 1024);
  float* att2 = (float*)take(sizeof(float) * 8 * 64 * 1024);
  float* mo1  = (float*)take(sizeof(float) * 8 * 64 * 1024);
  float* mo2  = (float*)take(sizeof(float) * 8 * 64 * 1024);
  float* s1   = (float*)take(sizeof(float) * 8 * 5 * 64);
  float* s2   = (float*)take(sizeof(float) * 8 * 5 * 64);

  mcn_prep<<<504, 256, 0, stream>>>(w2a, m_w1, m_w2, m_w3, Aw, Wq);
  mcn_conv3<<<dim3(224, 8), 128, 0, stream>>>(x1, w3, b3, g3, be3, m3, v3, y224h);
  mcn_conv2<<<dim3(16, 8), 256, 0, stream>>>(y224h, Aw, b2a, g2a, be2a, m2a, v2a, y64h);
  mcn_conv1<<<dim3(64, 8), 128, 0, stream>>>(x2, w2b, b2b, g2b, be2b, m2b, v2b, z64h);
  mcn_qkv<<<dim3(16, 8), 256, 0, stream>>>(y64h, Wq, m_b1, m_b2, m_b3, qkv1);
  mcn_qkv<<<dim3(16, 8), 256, 0, stream>>>(z64h, Wq, m_b1, m_b2, m_b3, qkv2);
  mcn_att<<<64, 256, 0, stream>>>(qkv1, m_wp, m_bp, att1);
  mcn_att<<<64, 256, 0, stream>>>(qkv2, m_wp, m_bp, att2);
  mcn_mact_conv<<<64, 256, 0, stream>>>(qkv1, att1, m_fc, m_depw, m_depb, m_rate1, m_rate2, mo1);
  mcn_mact_conv<<<64, 256, 0, stream>>>(qkv2, att2, m_fc, m_depw, m_depb, m_rate1, m_rate2, mo2);
  mcn_token<<<dim3(4, 8), 256, 0, stream>>>(mo1, token_wA, token_wV, pos, cls, s1);
  mcn_token<<<dim3(4, 8), 256, 0, stream>>>(mo2, token_wA, token_wV, pos, cls, s2);
  mcn_head<<<8, 128, 0, stream>>>(s1, s2, c_wq, c_wk, c_wv, c_wo, c_bo,
                                  ln_g, ln_b, head_w, head_b, (float*)d_out);
}
